// ContinualVQVAELayer_80607946211619
// MI455X (gfx1250) — compile-verified, matrix operand order checked by reading, not run on hardware
//
#include <hip/hip_runtime.h>

typedef __attribute__((ext_vector_type(16))) _Float16 v16h;
typedef __attribute__((ext_vector_type(8)))  _Float16 v8h;
typedef __attribute__((ext_vector_type(8)))  float    v8f;
typedef __attribute__((ext_vector_type(4)))  float    v4f;

#define DEV __device__ __forceinline__

// ---------------------------------------------------------------------------
// Fragment loaders.
// WMMA 16x16x32 f16 operand layout (wave32):
//   A: lane L holds row M = L%16.  Lanes 0-15: K = {0..7, 16..23},
//      lanes 16-31: K = {8..15, 24..31}.
//   B: lane L holds col N = L%16 with the same K striping.
// With A row-major [M][K] and B pre-transposed to [N][K], both fragments are
// two contiguous 16-byte loads per lane: base + (half?8:0) and +16 halves.
// ---------------------------------------------------------------------------
DEV v16h pack16(v8h lo, v8h hi) {
    v16h r;
#pragma unroll
    for (int i = 0; i < 8; ++i) { r[i] = lo[i]; r[i + 8] = hi[i]; }
    return r;
}

DEV v16h frag_f32(const float* __restrict__ base) {
    v4f a0 = *(const v4f*)(base);
    v4f a1 = *(const v4f*)(base + 4);
    v4f b0 = *(const v4f*)(base + 16);
    v4f b1 = *(const v4f*)(base + 20);
    v16h r;
#pragma unroll
    for (int i = 0; i < 4; ++i) {
        r[i]      = (_Float16)a0[i];
        r[i + 4]  = (_Float16)a1[i];
        r[i + 8]  = (_Float16)b0[i];
        r[i + 12] = (_Float16)b1[i];
    }
    return r;
}

DEV v16h frag_f16(const _Float16* __restrict__ base) {
    return pack16(*(const v8h*)(base), *(const v8h*)(base + 16));
}

// ---------------------------------------------------------------------------
// Generic fused GEMM: out = act(A[M,K] * Bt[N,K]^T + bias[N])
// Block = 256 threads = 8 waves; wave owns 16 rows x 64 cols (4 accum tiles).
// Bt / bias are small (<=512KB) -> L2-resident on MI455X (192MB L2).
// Software-pipelined: fragments for k-step i+1 are in flight while the
// WMMAs of k-step i issue, so waits become partial instead of loadcnt==0.
// ---------------------------------------------------------------------------
template <bool A_IS_F32>
__global__ __launch_bounds__(256) void gemm_bias_act(
    const void* __restrict__ Av,         // A  [M][K]  (f32 or f16)
    const _Float16* __restrict__ Bt,     // Bt [N][K]  f16 (pre-transposed)
    const float* __restrict__ bias,      // [N]
    _Float16* __restrict__ out16,        // optional f16 out [M][N]
    float* __restrict__ out32,           // optional f32 out [M][N]
    int M, int N, int K, int relu)
{
    const int lane = threadIdx.x & 31;
    const int wave = threadIdx.x >> 5;
    const int sub  = lane & 15;          // M (A) / N (B) index within tile
    const int half = lane >> 4;          // selects K sub-pattern
    const int rowbase = (blockIdx.x * 8 + wave) * 16;
    const int colbase = blockIdx.y * 64;
    if (rowbase >= M) return;

    const int koff = half * 8;
    const float*    af32 = (const float*)Av    + (size_t)(rowbase + sub) * K + koff;
    const _Float16* af16 = (const _Float16*)Av + (size_t)(rowbase + sub) * K + koff;
    const _Float16* bb   = Bt + (size_t)(colbase + sub) * K + koff;

    v8f acc[4] = {};
    const int nk = K >> 5;

    // prologue: k-step 0 in flight
    v16h a_cur = A_IS_F32 ? frag_f32(af32) : frag_f16(af16);
    v8h blo[4], bhi[4];
#pragma unroll
    for (int t = 0; t < 4; ++t) {
        const _Float16* p = bb + (size_t)t * 16 * K;
        blo[t] = *(const v8h*)(p);
        bhi[t] = *(const v8h*)(p + 16);
    }

#pragma unroll 2
    for (int ki = 0; ki < nk; ++ki) {
        const int k1 = (ki + 1 < nk) ? (ki + 1) * 32 : 0;   // wrap: valid, unused
        // prefetch k-step ki+1
        v16h a_nxt = A_IS_F32 ? frag_f32(af32 + k1) : frag_f16(af16 + k1);
        v8h nlo[4], nhi[4];
#pragma unroll
        for (int t = 0; t < 4; ++t) {
            const _Float16* p = bb + (size_t)t * 16 * K + k1;
            nlo[t] = *(const v8h*)(p);
            nhi[t] = *(const v8h*)(p + 16);
        }
        // consume k-step ki
#pragma unroll
        for (int t = 0; t < 4; ++t) {
            acc[t] = __builtin_amdgcn_wmma_f32_16x16x32_f16(
                false, a_cur, false, pack16(blo[t], bhi[t]),
                (short)0, acc[t], false, false);
        }
        a_cur = a_nxt;
#pragma unroll
        for (int t = 0; t < 4; ++t) { blo[t] = nlo[t]; bhi[t] = nhi[t]; }
    }

#pragma unroll
    for (int t = 0; t < 4; ++t) {
        const int col = colbase + t * 16 + sub;
        const float bv = bias[col];
#pragma unroll
        for (int r = 0; r < 8; ++r) {
            const int row = rowbase + r + half * 8;
            float v = acc[t][r] + bv;
            if (relu) v = v > 0.0f ? v : 0.0f;
            if (out16) out16[(size_t)row * N + col] = (_Float16)v;
            if (out32) out32[(size_t)row * N + col] = v;
        }
    }
}

// ---------------------------------------------------------------------------
// Fused VQ: distances via WMMA (z_e[16,128] x cb^T), running argmin,
// cross-lane reduce, codebook gather (z_q), indices, commitment-loss partial.
// Wave handles 16 rows against all 4096 codes; codebook f16 (1MB) is L2-hot.
// Code-tile loop is software-pipelined like the GEMM.
// ---------------------------------------------------------------------------
__global__ __launch_bounds__(256) void vq_argmin_gather(
    const float* __restrict__ ze32,      // [B][128]  f32 encoder output
    const _Float16* __restrict__ cb16,   // [4096][128] f16 (== Bt layout)
    const float* __restrict__ cb32,      // [4096][128] f32 original codebook
    const float* __restrict__ cnorm,     // [4096]  ||c||^2
    float* __restrict__ zq_out,          // [B][128] f32 (d_out region)
    int* __restrict__ idx_out,           // [B]
    float* __restrict__ loss_accum)      // scalar accumulator in ws
{
    const int lane = threadIdx.x & 31;
    const int wave = threadIdx.x >> 5;
    const int sub  = lane & 15;
    const int half = lane >> 4;
    const int rowbase = (blockIdx.x * 8 + wave) * 16;
    const int koff = half * 8;

    // Resident z_e fragments: K = 128 -> 4 fragments (32 VGPRs), f32->f16 inline.
    v16h a[4];
#pragma unroll
    for (int ks = 0; ks < 4; ++ks)
        a[ks] = frag_f32(ze32 + (size_t)(rowbase + sub) * 128 + ks * 32 + koff);

    float best[8];
    int   bidx[8];
#pragma unroll
    for (int r = 0; r < 8; ++r) { best[r] = 3.4e38f; bidx[r] = 0; }

    const _Float16* cb = cb16 + (size_t)sub * 128 + koff;  // this lane's column

    // prologue: tile n0 = 0 in flight
    v8h blo[4], bhi[4];
#pragma unroll
    for (int ks = 0; ks < 4; ++ks) {
        blo[ks] = *(const v8h*)(cb + ks * 32);
        bhi[ks] = *(const v8h*)(cb + ks * 32 + 16);
    }
    float cn = cnorm[sub];

#pragma unroll 2
    for (int n0 = 0; n0 < 4096; n0 += 16) {
        const int n1 = (n0 + 16 < 4096) ? n0 + 16 : 0;     // wrap: valid, unused
        // prefetch next code tile + its norm
        v8h nlo[4], nhi[4];
#pragma unroll
        for (int ks = 0; ks < 4; ++ks) {
            const _Float16* p = cb + (size_t)n1 * 128 + ks * 32;
            nlo[ks] = *(const v8h*)(p);
            nhi[ks] = *(const v8h*)(p + 16);
        }
        float cn_nxt = cnorm[n1 + sub];

        // consume current tile
        v8f dot = {};
#pragma unroll
        for (int ks = 0; ks < 4; ++ks) {
            dot = __builtin_amdgcn_wmma_f32_16x16x32_f16(
                false, a[ks], false, pack16(blo[ks], bhi[ks]),
                (short)0, dot, false, false);
        }
        const int id = n0 + sub;
#pragma unroll
        for (int r = 0; r < 8; ++r) {
            float v = cn - 2.0f * dot[r];   // ||z||^2 constant per row -> drop
            if (v < best[r]) { best[r] = v; bidx[r] = id; }
        }
        cn = cn_nxt;
#pragma unroll
        for (int ks = 0; ks < 4; ++ks) { blo[ks] = nlo[ks]; bhi[ks] = nhi[ks]; }
    }

    // Min-reduce across the 16 lanes of each half (columns live across lanes).
#pragma unroll
    for (int m = 1; m < 16; m <<= 1) {
#pragma unroll
        for (int r = 0; r < 8; ++r) {
            float ov = __shfl_xor(best[r], m, 32);
            int   oi = __shfl_xor(bidx[r], m, 32);
            if (ov < best[r] || (ov == best[r] && oi < bidx[r])) {
                best[r] = ov; bidx[r] = oi;
            }
        }
    }

    // Gather z_q (16 lanes cooperate: lane handles 8 of 128 dims) + loss.
    float lsum = 0.0f;
    const int d0 = sub * 8;
#pragma unroll
    for (int r = 0; r < 8; ++r) {
        const int row = rowbase + r + half * 8;
        const int id  = bidx[r];            // identical across the 16 lanes
        if (sub == 0) idx_out[row] = id;
        v4f c0 = *(const v4f*)(cb32 + (size_t)id * 128 + d0);
        v4f c1 = *(const v4f*)(cb32 + (size_t)id * 128 + d0 + 4);
        *(v4f*)(zq_out + (size_t)row * 128 + d0)     = c0;
        *(v4f*)(zq_out + (size_t)row * 128 + d0 + 4) = c1;
        v4f z0 = *(const v4f*)(ze32 + (size_t)row * 128 + d0);
        v4f z1 = *(const v4f*)(ze32 + (size_t)row * 128 + d0 + 4);
#pragma unroll
        for (int i = 0; i < 4; ++i) {
            float e0 = z0[i] - c0[i];
            float e1 = z1[i] - c1[i];
            lsum += e0 * e0 + e1 * e1;
        }
    }
#pragma unroll
    for (int m = 1; m < 32; m <<= 1) lsum += __shfl_xor(lsum, m, 32);
    if (lane == 0) atomicAdd(loss_accum, lsum);
}

// ---------------------------------------------------------------------------
// Pack / misc kernels
// ---------------------------------------------------------------------------
__global__ void transpose_cvt_f16(const float* __restrict__ src,  // [K][N]
                                  _Float16* __restrict__ dst,     // [N][K]
                                  int K, int N)
{
    int i = blockIdx.x * blockDim.x + threadIdx.x;
    if (i >= K * N) return;
    int k = i / N, n = i - k * N;
    dst[(size_t)n * K + k] = (_Float16)src[i];
}

__global__ void cvt_f16(const float* __restrict__ src, _Float16* __restrict__ dst, int n)
{
    int i = blockIdx.x * blockDim.x + threadIdx.x;
    if (i < n) dst[i] = (_Float16)src[i];
}

__global__ void codebook_norms(const float* __restrict__ cb, float* __restrict__ cnorm)
{
    int n = blockIdx.x * blockDim.x + threadIdx.x;
    if (n >= 4096) return;
    float s = 0.0f;
#pragma unroll 8
    for (int d = 0; d < 128; ++d) { float v = cb[(size_t)n * 128 + d]; s += v * v; }
    cnorm[n] = s;
}

__global__ void zero_scalar(float* p) { if (threadIdx.x == 0 && blockIdx.x == 0) *p = 0.0f; }

__global__ void finalize_loss(const float* __restrict__ acc, float* __restrict__ out)
{
    if (threadIdx.x == 0 && blockIdx.x == 0)
        *out = *acc * (1.0f / (65536.0f * 128.0f));
}

// ---------------------------------------------------------------------------
// Launch
// ---------------------------------------------------------------------------
extern "C" void kernel_launch(void* const* d_in, const int* in_sizes, int n_in,
                              void* d_out, int out_size, void* d_ws, size_t ws_size,
                              hipStream_t stream)
{
    constexpr int B = 65536, IN = 1024, H = 256, D = 128, CB = 4096;

    const float* x      = (const float*)d_in[0];
    const float* enc_w1 = (const float*)d_in[1];
    const float* enc_b1 = (const float*)d_in[2];
    const float* enc_w2 = (const float*)d_in[3];
    const float* enc_b2 = (const float*)d_in[4];
    const float* enc_w3 = (const float*)d_in[5];
    const float* enc_b3 = (const float*)d_in[6];
    const float* cb32   = (const float*)d_in[7];
    const float* dec_w1 = (const float*)d_in[8];
    const float* dec_b1 = (const float*)d_in[9];
    const float* dec_w2 = (const float*)d_in[10];
    const float* dec_b2 = (const float*)d_in[11];
    const float* dec_w3 = (const float*)d_in[12];
    const float* dec_b3 = (const float*)d_in[13];

    // d_out layout (flat, element index): x_recon | z_q | loss | indices
    float* out_f      = (float*)d_out;
    float* x_recon    = out_f;
    float* zq_out     = out_f + (size_t)B * IN;
    float* loss_out   = out_f + (size_t)B * IN + (size_t)B * D;
    int*   idx_out    = (int*)d_out + ((size_t)B * IN + (size_t)B * D + 1);

    // workspace layout (bytes)
    char* ws = (char*)d_ws;
    size_t off = 0;
    auto take = [&](size_t bytes) { void* p = ws + off; off = (off + bytes + 255) & ~(size_t)255; return p; };
    _Float16* w1t  = (_Float16*)take((size_t)IN * H * 2);   // enc_w1^T  [256][1024]
    _Float16* w2t  = (_Float16*)take((size_t)H * H * 2);    // enc_w2^T  [256][256]
    _Float16* w3t  = (_Float16*)take((size_t)H * D * 2);    // enc_w3^T  [128][256]
    _Float16* d1t  = (_Float16*)take((size_t)D * H * 2);    // dec_w1^T  [256][128]
    _Float16* d2t  = (_Float16*)take((size_t)H * H * 2);    // dec_w2^T  [256][256]
    _Float16* d3t  = (_Float16*)take((size_t)H * IN * 2);   // dec_w3^T  [1024][256]
    _Float16* cb16 = (_Float16*)take((size_t)CB * D * 2);   // codebook f16
    float*    cnrm = (float*)take((size_t)CB * 4);
    float*    lacc = (float*)take(4);
    float*    ze32 = (float*)take((size_t)B * D * 4);       // z_e f32
    _Float16* h1   = (_Float16*)take((size_t)B * H * 2);    // reused as g1
    _Float16* h2   = (_Float16*)take((size_t)B * H * 2);    // reused as g2
    (void)ws_size; (void)in_sizes; (void)n_in; (void)out_size;

    // --- pack weights (transpose + f16) ---
    auto tgrid = [](int k, int n) { return dim3((k * n + 255) / 256); };
    transpose_cvt_f16<<<tgrid(IN, H), 256, 0, stream>>>(enc_w1, w1t, IN, H);
    transpose_cvt_f16<<<tgrid(H, H),  256, 0, stream>>>(enc_w2, w2t, H, H);
    transpose_cvt_f16<<<tgrid(H, D),  256, 0, stream>>>(enc_w3, w3t, H, D);
    transpose_cvt_f16<<<tgrid(D, H),  256, 0, stream>>>(dec_w1, d1t, D, H);
    transpose_cvt_f16<<<tgrid(H, H),  256, 0, stream>>>(dec_w2, d2t, H, H);
    transpose_cvt_f16<<<tgrid(H, IN), 256, 0, stream>>>(dec_w3, d3t, H, IN);
    cvt_f16<<<(CB * D + 255) / 256, 256, 0, stream>>>(cb32, cb16, CB * D);
    codebook_norms<<<CB / 256, 256, 0, stream>>>(cb32, cnrm);
    zero_scalar<<<1, 64, 0, stream>>>(lacc);

    const int MB = B / 128;   // 512 row-blocks

    // --- encoder ---
    gemm_bias_act<true ><<<dim3(MB, H / 64), 256, 0, stream>>>(
        x, w1t, enc_b1, h1, nullptr, B, H, IN, 1);
    gemm_bias_act<false><<<dim3(MB, H / 64), 256, 0, stream>>>(
        h1, w2t, enc_b2, h2, nullptr, B, H, H, 1);
    gemm_bias_act<false><<<dim3(MB, D / 64), 256, 0, stream>>>(
        h2, w3t, enc_b3, nullptr, ze32, B, D, H, 0);

    // --- vector quantization (WMMA distances + argmin + gather + loss) ---
    vq_argmin_gather<<<dim3(MB), 256, 0, stream>>>(
        ze32, cb16, cb32, cnrm, zq_out, idx_out, lacc);

    // --- decoder (reads z_q f32 straight from d_out, converts inline) ---
    gemm_bias_act<true ><<<dim3(MB, H / 64), 256, 0, stream>>>(
        zq_out, d1t, dec_b1, h1, nullptr, B, H, D, 1);
    gemm_bias_act<false><<<dim3(MB, H / 64), 256, 0, stream>>>(
        h1, d2t, dec_b2, h2, nullptr, B, H, H, 1);
    gemm_bias_act<false><<<dim3(MB, IN / 64), 256, 0, stream>>>(
        h2, d3t, dec_b3, nullptr, x_recon, B, IN, H, 0);

    finalize_loss<<<1, 64, 0, stream>>>(lacc, loss_out);
}
